// VICRegLLoss_37572373905606
// MI455X (gfx1250) — compile-verified, matrix operand order checked by reading, not run on hardware
//
#include <hip/hip_runtime.h>

typedef _Float16 v16h __attribute__((ext_vector_type(16)));
typedef _Float16 v8h  __attribute__((ext_vector_type(8)));
typedef float    v8f  __attribute__((ext_vector_type(8)));
typedef float    v4f  __attribute__((ext_vector_type(4)));
typedef unsigned long long u64;

#define BATCH  64
#define NPTS   1024
#define DFEAT  512
#define GDIM   8192
#define KMATCH 10
#define TM     128
#define TK     32
#define LDH    40   // padded LDS row stride in halves (80B, keeps 16B alignment)

__device__ __forceinline__ u64 packKey(float d, unsigned int idx) {
  return ((u64)__float_as_uint(d) << 32) | (u64)idx;
}

// ---------------- init: nn buffers to ~0, accumulators to 0 ----------------
__global__ void k_init(u64* __restrict__ nnAll, float* __restrict__ Gbuf,
                       float* __restrict__ scalars) {
  int i = blockIdx.x * blockDim.x + threadIdx.x;
  if (i < 4 * BATCH * NPTS) nnAll[i] = ~0ull;
  if (i < 2 * 64 * 64)      Gbuf[i] = 0.f;
  if (i < 32)               scalars[i] = 0.f;
}

// ---------------- per-row squared norms of both map tensors ----------------
__global__ void k_norms(const float* __restrict__ x1, const float* __restrict__ x2,
                        float* __restrict__ norms) {
  int w = threadIdx.x >> 5, lane = threadIdx.x & 31;
  int row = blockIdx.x * 8 + w;                    // 0 .. 2*B*N-1
  const float* src = (row < BATCH * NPTS) ? x1 : x2;
  int r = row & (BATCH * NPTS - 1);
  const float* p = src + (size_t)r * DFEAT;
  float s = 0.f;
  for (int k = lane; k < DFEAT; k += 32) { float v = p[k]; s += v * v; }
#pragma unroll
  for (int off = 16; off; off >>= 1) s += __shfl_xor(s, off, 32);
  if (lane == 0) norms[row] = s;
}

// ---------------- location nearest-neighbor (one direction) ----------------
__global__ void k_loc_nn(const float* __restrict__ aloc, const float* __restrict__ bloc,
                         u64* __restrict__ nnout) {
  __shared__ float sbx[NPTS], sby[NPTS], snb[NPTS];
  int b = blockIdx.x, tid = threadIdx.x;
  const float* bp = bloc + (size_t)b * NPTS * 2;
  for (int i = tid; i < NPTS; i += 256) {
    float bx = bp[i * 2], by = bp[i * 2 + 1];
    sbx[i] = bx; sby[i] = by; snb[i] = bx * bx + by * by;
  }
  __syncthreads();
  int n = blockIdx.y * 256 + tid;
  const float* ap = aloc + ((size_t)b * NPTS + n) * 2;
  float ax = ap[0], ay = ap[1];
  float sa = ax * ax + ay * ay;
  u64 best = ~0ull;
  for (int m = 0; m < NPTS; ++m) {
    float d2 = fmaxf(sa + snb[m] - 2.f * (ax * sbx[m] + ay * sby[m]), 0.f);
    u64 key = packKey(d2, (unsigned)m);
    if (key < best) best = key;
  }
  nnout[(size_t)b * NPTS + n] = best;
}

// ---------------- WMMA fragment load (16-bit A layout from ISA doc) --------
__device__ __forceinline__ v16h loadFrag(const _Float16* base, int lane) {
  const _Float16* p = base + (lane & 15) * LDH + ((lane >> 4) << 3);
  v8h lo = *reinterpret_cast<const v8h*>(p);        // K = kb .. kb+7
  v8h hi = *reinterpret_cast<const v8h*>(p + 16);   // K = kb+16 .. kb+23
  v16h f;
#pragma unroll
  for (int e = 0; e < 8; ++e) { f[e] = lo[e]; f[e + 8] = hi[e]; }
  return f;
}

// ------- feature cdist: WMMA tile + fused row/col argmin via atomics -------
// Software-pipelined: chunk k+1 global loads are issued before chunk k's
// WMMAs so the s_wait_loadcnt lands one iteration later (latency hidden).
__global__ void __launch_bounds__(256)
k_feat(const float* __restrict__ x1m, const float* __restrict__ x2m,
       const float* __restrict__ norms, u64* __restrict__ nnrow, u64* __restrict__ nncol) {
  __shared__ _Float16 As[TM * LDH];
  __shared__ _Float16 Bs[TM * LDH];
  __shared__ float raS[TM], rbS[TM];

  const int b  = blockIdx.x;
  const int r0 = blockIdx.y * TM;
  const int c0 = blockIdx.z * TM;
  const int tid = threadIdx.x;
  const int lane = tid & 31;
  const int w = tid >> 5;
  const int wr = w >> 1;  // 0..3 -> 32-row strip
  const int wc = w & 1;   // 0..1 -> 64-col strip

  if (tid < 128) raS[tid] = norms[b * NPTS + r0 + tid];
  else           rbS[tid - 128] = norms[BATCH * NPTS + b * NPTS + c0 + (tid - 128)];

  const float* Abase = x1m + ((size_t)b * NPTS + r0) * DFEAT;
  const float* Bbase = x2m + ((size_t)b * NPTS + c0) * DFEAT;

  // per-thread staging slots: slot s = tid + q*256 -> row = s>>3, kq = (s&7)*4
  const int srow[4] = { (tid + 0) >> 3, (tid + 256) >> 3, (tid + 512) >> 3, (tid + 768) >> 3 };
  const int skq[4]  = { ((tid + 0) & 7) << 2, ((tid + 256) & 7) << 2,
                        ((tid + 512) & 7) << 2, ((tid + 768) & 7) << 2 };

  v4f pa[4], pb[4];
#pragma unroll
  for (int q = 0; q < 4; ++q) {                      // prologue: chunk 0
    pa[q] = *reinterpret_cast<const v4f*>(Abase + (size_t)srow[q] * DFEAT + skq[q]);
    pb[q] = *reinterpret_cast<const v4f*>(Bbase + (size_t)srow[q] * DFEAT + skq[q]);
  }

  const v8f vz = {0.f, 0.f, 0.f, 0.f, 0.f, 0.f, 0.f, 0.f};
  v8f acc[2][4];
#pragma unroll
  for (int i = 0; i < 2; ++i)
#pragma unroll
    for (int j = 0; j < 4; ++j) acc[i][j] = vz;

#pragma unroll 1
  for (int kc = 0; kc < DFEAT; kc += TK) {
    __syncthreads();                                 // prev chunk fully consumed
    // convert+store the prefetched chunk (fp32 -> f16)
#pragma unroll
    for (int q = 0; q < 4; ++q) {
      union { _Float16 h[4]; u64 u; } ua, ub;
#pragma unroll
      for (int e = 0; e < 4; ++e) { ua.h[e] = (_Float16)pa[q][e]; ub.h[e] = (_Float16)pb[q][e]; }
      *reinterpret_cast<u64*>(&As[srow[q] * LDH + skq[q]]) = ua.u;
      *reinterpret_cast<u64*>(&Bs[srow[q] * LDH + skq[q]]) = ub.u;
    }
    __syncthreads();
    // prefetch next chunk while this chunk's WMMAs execute
    if (kc + TK < DFEAT) {
#pragma unroll
      for (int q = 0; q < 4; ++q) {
        pa[q] = *reinterpret_cast<const v4f*>(Abase + (size_t)srow[q] * DFEAT + kc + TK + skq[q]);
        pb[q] = *reinterpret_cast<const v4f*>(Bbase + (size_t)srow[q] * DFEAT + kc + TK + skq[q]);
      }
    }
    v16h af[2], bf[4];
#pragma unroll
    for (int i = 0; i < 2; ++i) af[i] = loadFrag(&As[(wr * 32 + i * 16) * LDH], lane);
#pragma unroll
    for (int j = 0; j < 4; ++j) bf[j] = loadFrag(&Bs[(wc * 64 + j * 16) * LDH], lane);
#pragma unroll
    for (int i = 0; i < 2; ++i)
#pragma unroll
      for (int j = 0; j < 4; ++j)
        acc[i][j] = __builtin_amdgcn_wmma_f32_16x16x32_f16(
            false, af[i], false, bf[j], (short)0, acc[i][j], false, false);
  }
  __syncthreads();

  const int h  = lane >> 4;   // C layout: VGPR v holds M=v (lanes<16) / v+8
  const int nl = lane & 15;   // N = lane%16

  // --- row minima (min over this tile's 128 columns) ---
#pragma unroll
  for (int i = 0; i < 2; ++i) {
#pragma unroll
    for (int v = 0; v < 8; ++v) {
      int lrow = wr * 32 + i * 16 + v + 8 * h;
      float ra = raS[lrow];
      float bv = 3.4e38f; int bi = 0x7FFFFFFF;
#pragma unroll
      for (int j = 0; j < 4; ++j) {
        int lcol = wc * 64 + j * 16 + nl;
        float d = fmaxf(ra + rbS[lcol] - 2.f * acc[i][j][v], 0.f);
        if (d < bv) { bv = d; bi = c0 + lcol; }   // increasing col -> first wins
      }
#pragma unroll
      for (int off = 1; off < 16; off <<= 1) {
        float ov = __shfl_xor(bv, off, 32);
        int   oi = __shfl_xor(bi, off, 32);
        if (ov < bv || (ov == bv && oi < bi)) { bv = ov; bi = oi; }
      }
      if (nl == 0)
        atomicMin(&nnrow[(size_t)b * NPTS + r0 + lrow], packKey(bv, (unsigned)bi));
    }
  }
  // --- column minima (min over this tile's 128 rows) ---
#pragma unroll
  for (int j = 0; j < 4; ++j) {
    int lcol = wc * 64 + j * 16 + nl;
    float rb = rbS[lcol];
    float bv = 3.4e38f; int bi = 0x7FFFFFFF;
#pragma unroll
    for (int i = 0; i < 2; ++i)
#pragma unroll
      for (int v = 0; v < 8; ++v) {
        int lrow = wr * 32 + i * 16 + v + 8 * h;
        float d = fmaxf(raS[lrow] + rb - 2.f * acc[i][j][v], 0.f);
        int gi = r0 + lrow;
        if (d < bv || (d == bv && gi < bi)) { bv = d; bi = gi; }
      }
    {
      float ov = __shfl_xor(bv, 16, 32);
      int   oi = __shfl_xor(bi, 16, 32);
      if (ov < bv || (ov == bv && oi < bi)) { bv = ov; bi = oi; }
    }
    if (h == 0)
      atomicMin(&nncol[(size_t)b * NPTS + c0 + lcol], packKey(bv, (unsigned)bi));
  }
}

// ------- global vicreg: per-column mean/var/relu + diag(q) + inv sum -------
__global__ void k_gstats(const float* __restrict__ x1g, const float* __restrict__ x2g,
                         float* __restrict__ colmean, float* __restrict__ scalars) {
  int m = blockIdx.y;
  int j = blockIdx.x * 256 + threadIdx.x;
  const float* g = m ? x2g : x1g;
  float s = 0.f, s2 = 0.f, sd = 0.f;
  for (int n = 0; n < BATCH; ++n) {
    float v = g[(size_t)n * GDIM + j];
    s += v; s2 += v * v;
    if (m == 0) { float d = v - x2g[(size_t)n * GDIM + j]; sd += d * d; }
  }
  float mu  = s * (1.f / BATCH);
  float q   = fmaxf(s2 - BATCH * mu * mu, 0.f);   // sum (x-mu)^2  (= diag of Z^T Z)
  float var = q * (1.f / (BATCH - 1));
  float rl  = fmaxf(1.f - sqrtf(var), 0.f);
  colmean[(size_t)m * GDIM + j] = mu;
  float qq = q * q;
#pragma unroll
  for (int off = 16; off; off >>= 1) {
    rl += __shfl_xor(rl, off, 32);
    qq += __shfl_xor(qq, off, 32);
    sd += __shfl_xor(sd, off, 32);
  }
  __shared__ float sr[8], sq[8], ss[8];
  int w = threadIdx.x >> 5, lane = threadIdx.x & 31;
  if (lane == 0) { sr[w] = rl; sq[w] = qq; ss[w] = sd; }
  __syncthreads();
  if (threadIdx.x == 0) {
    float a = 0, bq = 0, c = 0;
    for (int i = 0; i < 8; ++i) { a += sr[i]; bq += sq[i]; c += ss[i]; }
    atomicAdd(&scalars[1 + m], a);   // Wvar
    atomicAdd(&scalars[3 + m], bq);  // sum q^2 (diag correction)
    if (m == 0) atomicAdd(&scalars[0], c);  // inv sum
  }
}

// ------- 64x64 Gram of centered Z (per 128-column chunk, atomic merge) -----
__global__ void k_gram(const float* __restrict__ x1g, const float* __restrict__ x2g,
                       const float* __restrict__ colmean, float* __restrict__ Gbuf) {
  __shared__ float Z[64 * 129];
  int m  = blockIdx.y;
  int c0 = blockIdx.x * 128;
  const float* g  = m ? x2g : x1g;
  const float* mu = colmean + (size_t)m * GDIM + c0;
  int tid = threadIdx.x;
  for (int idx = tid; idx < 64 * 128; idx += 256) {
    int n = idx >> 7, j = idx & 127;
    Z[n * 129 + j] = g[(size_t)n * GDIM + c0 + j] - mu[j];
  }
  __syncthreads();
  for (int p = 0; p < 16; ++p) {
    int pair = tid + p * 256;
    int n = pair >> 6, mm = pair & 63;
    float acc = 0.f;
    for (int j = 0; j < 128; ++j) acc += Z[n * 129 + j] * Z[mm * 129 + j];
    atomicAdd(&Gbuf[m * 4096 + pair], acc);
  }
}

// ------- top-10 smallest NN distances per batch + local inv/var stats ------
__global__ void k_topk(const u64* __restrict__ nn, const float* __restrict__ im,
                       const float* __restrict__ cm, float* __restrict__ scalars, int combo) {
  __shared__ u64 keys[NPTS];
  __shared__ unsigned int cand[NPTS];
  __shared__ u64 wred[8];
  __shared__ u64 winner;
  __shared__ float fi_s[KMATCH], fc_s[KMATCH];
  int b = blockIdx.x, tid = threadIdx.x;
  for (int t = tid; t < NPTS; t += 256) {
    u64 p = nn[(size_t)b * NPTS + t];
    cand[t] = (unsigned int)(p & 0xFFFFFFFFull);
    keys[t] = (p & 0xFFFFFFFF00000000ull) | (u64)t;  // (dist, element idx)
  }
  __syncthreads();
  int w = tid >> 5, lane = tid & 31;
  for (int k = 0; k < KMATCH; ++k) {
    u64 mn = ~0ull;
    for (int t = tid; t < NPTS; t += 256) { u64 v = keys[t]; mn = v < mn ? v : mn; }
#pragma unroll
    for (int off = 16; off; off >>= 1) {
      u64 o = __shfl_xor(mn, off, 32);
      if (o < mn) mn = o;
    }
    if (lane == 0) wred[w] = mn;
    __syncthreads();
    if (tid == 0) {
      u64 mm = wred[0];
      for (int i = 1; i < 8; ++i) if (wred[i] < mm) mm = wred[i];
      winner = mm;
    }
    __syncthreads();
    for (int t = tid; t < NPTS; t += 256) if (keys[t] == winner) keys[t] = ~0ull;
    if (tid == 0) {
      unsigned int e = (unsigned int)(winner & 0xFFFFFFFFull);
      unsigned int c = cand[e];
      fi_s[k] = im[((size_t)b * NPTS + e) * DFEAT];   // channel 0 only (faithful)
      fc_s[k] = cm[((size_t)b * NPTS + c) * DFEAT];
    }
    __syncthreads();
  }
  if (tid == 0) {
    float sdiff = 0, mi = 0, mc2 = 0;
    for (int k = 0; k < KMATCH; ++k) {
      float d = fi_s[k] - fc_s[k];
      sdiff += d * d; mi += fi_s[k]; mc2 += fc_s[k];
    }
    mi *= (1.f / KMATCH); mc2 *= (1.f / KMATCH);
    float vi = 0, vc = 0;
    for (int k = 0; k < KMATCH; ++k) {
      float a = fi_s[k] - mi;  vi += a * a;
      float b2 = fc_s[k] - mc2; vc += b2 * b2;
    }
    vi *= (1.f / (KMATCH - 1)); vc *= (1.f / (KMATCH - 1));
    atomicAdd(&scalars[5 + combo],  sdiff);
    atomicAdd(&scalars[9 + combo],  fmaxf(1.f - sqrtf(vi), 0.f));
    atomicAdd(&scalars[13 + combo], fmaxf(1.f - sqrtf(vc), 0.f));
  }
}

// ---------------- finalize all 13 outputs ----------------
__global__ void k_final(const float* __restrict__ scalars, const float* __restrict__ Gbuf,
                        float* __restrict__ out) {
  __shared__ float red[256];
  __shared__ float gss[2];
  int tid = threadIdx.x;
  for (int m = 0; m < 2; ++m) {
    float s = 0.f;
    for (int i = tid; i < 4096; i += 256) { float v = Gbuf[m * 4096 + i]; s += v * v; }
    red[tid] = s; __syncthreads();
    for (int st = 128; st; st >>= 1) { if (tid < st) red[tid] += red[tid + st]; __syncthreads(); }
    if (tid == 0) gss[m] = red[0];
    __syncthreads();
  }
  if (tid == 0) {
    const float INVC = 25.f, VARC = 15.f, COVC = 1.f, ALPHA_ = 0.75f;
    float inv_g = INVC * scalars[0] / (float)(BATCH * GDIM);
    float var_g = VARC * (scalars[1] + scalars[2]) / (2.f * GDIM);
    float denom = 63.f * 63.f * (float)GDIM;
    float cov_g = COVC * 0.5f * ((gss[0] - scalars[3]) / denom + (gss[1] - scalars[4]) / denom);
    float loss_g = inv_g + var_g + cov_g;
    float li[4], lv[4], ll[4];
    for (int c = 0; c < 4; ++c) {
      li[c] = INVC * scalars[5 + c] / (float)(BATCH * KMATCH);
      lv[c] = VARC * (scalars[9 + c] / BATCH + scalars[13 + c] / BATCH) * 0.5f;
      ll[c] = li[c] + lv[c];   // local cov == 0 (D==1 -> off-diag mask empty)
    }
    float loc_inv = (li[0] + li[1]) * 0.5f, loc_var = (lv[0] + lv[1]) * 0.5f;
    float loc_loss = (ll[0] + ll[1]) * 0.5f;
    float ft_inv = (li[2] + li[3]) * 0.5f, ft_var = (lv[2] + lv[3]) * 0.5f;
    float ft_loss = (ll[2] + ll[3]) * 0.5f;
    float total = ALPHA_ * loss_g + (1.f - ALPHA_) * (ft_loss + loc_loss) * 0.5f;
    out[0] = total;
    out[1] = inv_g;  out[2] = var_g;  out[3] = cov_g; out[4] = loss_g;
    out[5] = loc_inv; out[6] = loc_var; out[7] = 0.f; out[8] = loc_loss;
    out[9] = ft_inv;  out[10] = ft_var; out[11] = 0.f; out[12] = ft_loss;
  }
}

extern "C" void kernel_launch(void* const* d_in, const int* in_sizes, int n_in,
                              void* d_out, int out_size, void* d_ws, size_t ws_size,
                              hipStream_t stream) {
  (void)in_sizes; (void)n_in; (void)out_size; (void)ws_size;
  const float* x1m = (const float*)d_in[0];
  const float* x2m = (const float*)d_in[1];
  const float* x1g = (const float*)d_in[2];
  const float* x2g = (const float*)d_in[3];
  const float* l1  = (const float*)d_in[4];
  const float* l2  = (const float*)d_in[5];
  float* out = (float*)d_out;

  char* ws = (char*)d_ws;
  u64* nnl1 = (u64*)ws;                 ws += (size_t)BATCH * NPTS * sizeof(u64);
  u64* nnl2 = (u64*)ws;                 ws += (size_t)BATCH * NPTS * sizeof(u64);
  u64* nnf1 = (u64*)ws;                 ws += (size_t)BATCH * NPTS * sizeof(u64);
  u64* nnf2 = (u64*)ws;                 ws += (size_t)BATCH * NPTS * sizeof(u64);
  float* norms   = (float*)ws;          ws += (size_t)2 * BATCH * NPTS * sizeof(float);
  float* colmean = (float*)ws;          ws += (size_t)2 * GDIM * sizeof(float);
  float* Gbuf    = (float*)ws;          ws += (size_t)2 * 64 * 64 * sizeof(float);
  float* scalars = (float*)ws;          ws += 32 * sizeof(float);

  k_init<<<dim3(1024), dim3(256), 0, stream>>>(nnl1, Gbuf, scalars);
  k_norms<<<dim3((2 * BATCH * NPTS) / 8), dim3(256), 0, stream>>>(x1m, x2m, norms);
  k_loc_nn<<<dim3(BATCH, 4), dim3(256), 0, stream>>>(l1, l2, nnl1);
  k_loc_nn<<<dim3(BATCH, 4), dim3(256), 0, stream>>>(l2, l1, nnl2);
  k_feat<<<dim3(BATCH, 8, 8), dim3(256), 0, stream>>>(x1m, x2m, norms, nnf1, nnf2);
  k_gstats<<<dim3(GDIM / 256, 2), dim3(256), 0, stream>>>(x1g, x2g, colmean, scalars);
  k_gram<<<dim3(GDIM / 128, 2), dim3(256), 0, stream>>>(x1g, x2g, colmean, Gbuf);
  k_topk<<<dim3(BATCH), dim3(256), 0, stream>>>(nnl1, x1m, x2m, scalars, 0);
  k_topk<<<dim3(BATCH), dim3(256), 0, stream>>>(nnl2, x2m, x1m, scalars, 1);
  k_topk<<<dim3(BATCH), dim3(256), 0, stream>>>(nnf1, x1m, x2m, scalars, 2);
  k_topk<<<dim3(BATCH), dim3(256), 0, stream>>>(nnf2, x2m, x1m, scalars, 3);
  k_final<<<dim3(1), dim3(256), 0, stream>>>(scalars, Gbuf, out);
}
